// VanillaRNN_36335423324380
// MI455X (gfx1250) — matrix operand order repaired and checked by reading, not compiled, not run on hardware
//
#include <hip/hip_runtime.h>
#include <math.h>

#define Bsz 256
#define Tsz 512
#define Dsz 128
#define Hsz 256
#define Osz 128

typedef __attribute__((ext_vector_type(16))) __bf16 v16bf;
typedef __attribute__((ext_vector_type(8)))  float  v8f;

union AFrag { uint4 q[2]; v16bf v; };

__device__ __forceinline__ v8f wmma_bf16(v16bf a, v16bf b, v8f c) {
  // D = A(16x32 bf16) x B(32x16 bf16) + C(16x16 f32)
  return __builtin_amdgcn_wmma_f32_16x16x32_bf16(false, a, false, b, (short)0, c,
                                                 false, false);
}

__device__ __forceinline__ unsigned short f2bf(float f) {
  __bf16 h = (__bf16)f;
  return __builtin_bit_cast(unsigned short, h);
}

// A-fragment (16x32 bf16, ISA 7.12.2 layout) from an LDS tile.
// Row-major tile, strideH bf16/row (strideH % 8 == 0 so rows stay 16B aligned).
// half=0: VGPR0-3 <- K=0..7, VGPR4-7 <- K=16..23 ; half=1: +8.
__device__ __forceinline__ v16bf load_a_lds(const unsigned short* tile, int strideH,
                                            int lane, int kBase) {
  const int m = lane & 15, half = lane >> 4;
  const unsigned short* p = tile + m * strideH + kBase + half * 8;
  AFrag f;
  f.q[0] = *(const uint4*)(p);        // K = kBase + half*8 .. +7
  f.q[1] = *(const uint4*)(p + 16);   // K = kBase + 16 + half*8 .. +7
  return f.v;
}

// B-fragment (32x16 bf16): B[k][n] = W[n][k], W row-major [N][ldk] f32 in global.
// Lane layout: N = n0 + lane%16, element e -> K = k0 + half*8 + (e < 8 ? e : e + 8).
__device__ __forceinline__ v16bf load_b_global(const float* __restrict__ W, int ldk,
                                               int n0, int k0, int lane) {
  const int n = n0 + (lane & 15), half = lane >> 4;
  const float* src = W + (size_t)n * ldk + k0 + half * 8;
  v16bf v;
#pragma unroll
  for (int e = 0; e < 16; ++e) v[e] = (__bf16)src[e < 8 ? e : e + 8];
  return v;
}

__device__ __forceinline__ float fast_tanh(float x) {
#if defined(__AMDGCN__) && __has_builtin(__builtin_amdgcn_tanhf)
  return __builtin_amdgcn_tanhf(x);   // CDNA5 V_TANH_F32
#else
  return tanhf(x);
#endif
}

// ---------------------------------------------------------------------------
// Kernel 1: xw[b,t,:] = x[b,t,:] @ W_hx^T + b_hx   (M=131072, N=256, K=128)
// Block tile: 64(M) x 256(N). 8 waves; wave w owns N-tiles {2w, 2w+1};
// W_hx B-fragments live in registers, x tile staged to LDS as bf16.
// x is streamed once -> non-temporal loads (keep L2 free for xw residency).
// ---------------------------------------------------------------------------
#define PSTR 136  // 128 + 8 pad (bank-conflict avoidance; keeps 16B row align)

__global__ __launch_bounds__(256) void rnn_proj(const float* __restrict__ x,
                                                const float* __restrict__ Whx,
                                                const float* __restrict__ bhx,
                                                float* __restrict__ xw) {
  __shared__ alignas(16) unsigned short sx[64 * PSTR];
  const int tid = threadIdx.x, lane = tid & 31, wave = tid >> 5;
  const int m = lane & 15, half = lane >> 4;
  const size_t mBase = (size_t)blockIdx.x * 64;

  v16bf Bf[2][4];
#pragma unroll
  for (int nt = 0; nt < 2; ++nt)
#pragma unroll
    for (int ks = 0; ks < 4; ++ks)
      Bf[nt][ks] = load_b_global(Whx, Dsz, (2 * wave + nt) * 16, ks * 32, lane);
  const float bias[2] = { bhx[(2 * wave + 0) * 16 + m],
                          bhx[(2 * wave + 1) * 16 + m] };

  for (int i = tid; i < 64 * Dsz; i += 256) {
    int r = i >> 7, c = i & (Dsz - 1);
    sx[r * PSTR + c] = f2bf(__builtin_nontemporal_load(&x[(mBase + r) * Dsz + c]));
  }
  __syncthreads();

  v8f acc[4][2];
#pragma unroll
  for (int mt = 0; mt < 4; ++mt)
#pragma unroll
    for (int nt = 0; nt < 2; ++nt)
#pragma unroll
      for (int r = 0; r < 8; ++r) acc[mt][nt][r] = bias[nt];

#pragma unroll
  for (int ks = 0; ks < 4; ++ks)
#pragma unroll
    for (int mt = 0; mt < 4; ++mt) {
      v16bf a = load_a_lds(sx + mt * 16 * PSTR, PSTR, lane, ks * 32);
      acc[mt][0] = wmma_bf16(a, Bf[0][ks], acc[mt][0]);
      acc[mt][1] = wmma_bf16(a, Bf[1][ks], acc[mt][1]);
    }

#pragma unroll
  for (int mt = 0; mt < 4; ++mt)
#pragma unroll
    for (int nt = 0; nt < 2; ++nt) {
      const int n = (2 * wave + nt) * 16 + m;
#pragma unroll
      for (int r = 0; r < 8; ++r)
        xw[(mBase + mt * 16 + half * 8 + r) * Hsz + n] = acc[mt][nt][r];
    }
}

// ---------------------------------------------------------------------------
// Kernel 2: persistent batch-sliced scan + fused output projection.
// 16 blocks x 16 batch rows. h slice lives in LDS (bf16, double buffered);
// W_hh lives in registers (16 B-frags per wave). One barrier per timestep.
// Two software pipelines keep latency off the sequential chain:
//   - xw[.,t+1,.] global loads issued at the top of step t,
//   - A-fragment ds_loads for kstep k+1 issued before the WMMAs of kstep k.
// ---------------------------------------------------------------------------
#define SSTR 264  // 256 + 8 pad

__global__ __launch_bounds__(256) void rnn_scan(const float* __restrict__ xw,
                                                const float* __restrict__ Whh,
                                                const float* __restrict__ bhh,
                                                const float* __restrict__ Wph,
                                                const float* __restrict__ bph,
                                                float* __restrict__ out) {
  __shared__ alignas(16) unsigned short hbuf[2][16 * SSTR];
  const int tid = threadIdx.x, lane = tid & 31, wave = tid >> 5;
  const int m = lane & 15, half = lane >> 4;
  const int bBase = blockIdx.x * 16;
  const int n0 = (2 * wave + 0) * 16 + m;
  const int n1 = (2 * wave + 1) * 16 + m;

  // Register-resident W_hh: wave w owns N-tiles {2w, 2w+1}, full K=256.
  v16bf Bf[2][8];
#pragma unroll
  for (int nt = 0; nt < 2; ++nt)
    for (int ks = 0; ks < 8; ++ks)
      Bf[nt][ks] = load_b_global(Whh, Hsz, (2 * wave + nt) * 16, ks * 32, lane);
  const float bias0 = bhh[n0];
  const float bias1 = bhh[n1];

  // Per-lane base pointers for the xw stream (row = bBase + half*8 + r).
  const float* xrow[8];
#pragma unroll
  for (int r = 0; r < 8; ++r)
    xrow[r] = xw + (size_t)(bBase + half * 8 + r) * Tsz * Hsz;

  for (int i = tid; i < 16 * SSTR; i += 256) hbuf[0][i] = 0;  // h0 = 0

  // Prefetch xw for t = 0.
  float nx0[8], nx1[8];
#pragma unroll
  for (int r = 0; r < 8; ++r) {
    nx0[r] = xrow[r][n0];
    nx1[r] = xrow[r][n1];
  }
  __syncthreads();

  int cur = 0;
  for (int t = 0; t < Tsz; ++t) {
    // C-init from prefetched xw + b_hh  (C layout: M = r + half*8, N = lane%16)
    v8f c0, c1;
#pragma unroll
    for (int r = 0; r < 8; ++r) {
      c0[r] = nx0[r] + bias0;
      c1[r] = nx1[r] + bias1;
    }
    // Issue next step's xw loads now; they complete under the WMMA work below.
    if (t + 1 < Tsz) {
      const size_t off = (size_t)(t + 1) * Hsz;
#pragma unroll
      for (int r = 0; r < 8; ++r) {
        nx0[r] = xrow[r][off + n0];
        nx1[r] = xrow[r][off + n1];
      }
    }
    // K loop with A-fragment double buffering (load k+1 before WMMAs of k).
    {
      v16bf a_cur = load_a_lds(hbuf[cur], SSTR, lane, 0);
#pragma unroll
      for (int ks = 0; ks < 8; ++ks) {
        v16bf a_next = a_cur;
        if (ks < 7) a_next = load_a_lds(hbuf[cur], SSTR, lane, (ks + 1) * 32);
        c0 = wmma_bf16(a_cur, Bf[0][ks], c0);
        c1 = wmma_bf16(a_cur, Bf[1][ks], c1);
        a_cur = a_next;
      }
    }
    unsigned short* hn = hbuf[cur ^ 1];
#pragma unroll
    for (int r = 0; r < 8; ++r) {
      const int row = half * 8 + r;
      hn[row * SSTR + n0] = f2bf(fast_tanh(c0[r]));
      hn[row * SSTR + n1] = f2bf(fast_tanh(c1[r]));
    }
    __syncthreads();
    cur ^= 1;
  }

  // Fused output projection: out[16 x 128] = h_T @ W_ph^T + b_ph.
  // Wave w owns N-tile w (O = 128 -> 8 tiles, one per wave).
  v16bf Bp[8];
#pragma unroll
  for (int ks = 0; ks < 8; ++ks)
    Bp[ks] = load_b_global(Wph, Hsz, wave * 16, ks * 32, lane);
  const float bp = bph[wave * 16 + m];
  v8f c;
#pragma unroll
  for (int r = 0; r < 8; ++r) c[r] = bp;
  {
    v16bf a_cur = load_a_lds(hbuf[cur], SSTR, lane, 0);
#pragma unroll
    for (int ks = 0; ks < 8; ++ks) {
      v16bf a_next = a_cur;
      if (ks < 7) a_next = load_a_lds(hbuf[cur], SSTR, lane, (ks + 1) * 32);
      c = wmma_bf16(a_cur, Bp[ks], c);
      a_cur = a_next;
    }
  }
#pragma unroll
  for (int r = 0; r < 8; ++r)
    out[(size_t)(bBase + half * 8 + r) * Osz + wave * 16 + m] = c[r];
}

// ---------------------------------------------------------------------------

extern "C" void kernel_launch(void* const* d_in, const int* in_sizes, int n_in,
                              void* d_out, int out_size, void* d_ws, size_t ws_size,
                              hipStream_t stream) {
  (void)in_sizes; (void)n_in; (void)out_size; (void)ws_size;
  const float* x   = (const float*)d_in[0];
  const float* Whx = (const float*)d_in[1];
  const float* bhx = (const float*)d_in[2];
  const float* Whh = (const float*)d_in[3];
  const float* bhh = (const float*)d_in[4];
  const float* Wph = (const float*)d_in[5];
  const float* bph = (const float*)d_in[6];
  float* out = (float*)d_out;
  float* xwq = (float*)d_ws;  // [B*T, H] f32 scratch = 128 MiB

  rnn_proj<<<(Bsz * Tsz) / 64, 256, 0, stream>>>(x, Whx, bhx, xwq);
  rnn_scan<<<Bsz / 16, 256, 0, stream>>>(xwq, Whh, bhh, Wph, bph, out);
}
